// GCN_33440615366929
// MI455X (gfx1250) — matrix operand order repaired and checked by reading, not compiled, not run on hardware
//
#include <hip/hip_runtime.h>
#include <hip/hip_bf16.h>

typedef __attribute__((ext_vector_type(16))) _Float16 v16h;
typedef __attribute__((ext_vector_type(8)))  _Float16 v8h;
typedef __attribute__((ext_vector_type(8)))  float    v8f;

// ---------------------------------------------------------------------------
// Degree / normalization kernels
// ---------------------------------------------------------------------------
__global__ void deg_init_kernel(float* __restrict__ deg, int N) {
    int i = blockIdx.x * blockDim.x + threadIdx.x;
    if (i < N) deg[i] = 1.0f;   // self-loop contributes 1 to every node's degree
}

__global__ void deg_accum_kernel(const int* __restrict__ dst, float* __restrict__ deg, int E) {
    int i = blockIdx.x * blockDim.x + threadIdx.x;
    if (i < E) unsafeAtomicAdd(&deg[dst[i]], 1.0f);   // hw global_atomic_add_f32
}

__global__ void dinv_kernel(float* __restrict__ deg, int N) {
    int i = blockIdx.x * blockDim.x + threadIdx.x;
    if (i < N) {
        float d = deg[i];
        deg[i] = (d > 0.0f) ? rsqrtf(d) : 0.0f;       // in-place deg -> dinv
    }
}

// ---------------------------------------------------------------------------
// WMMA GEMM:  G[row] = (X[row] @ W) * dinv[row];  ACC[row] = G[row] (self loop)
// One wave computes a 16x16 output tile with v_wmma_f32_16x16x32_f16.
// W is staged in LDS transposed (F x K) as f16 so B fragments are contiguous.
// All K-step B fragments are loaded before the WMMA chain so ds loads pipeline.
// ---------------------------------------------------------------------------
template<int K, int F, bool GATHER>
__global__ void __launch_bounds__(128)
gcn_gemm_kernel(const float* __restrict__ X, const float* __restrict__ W,
                const int* __restrict__ nodes, const float* __restrict__ dinv,
                float* __restrict__ G, float* __restrict__ ACC, int N)
{
    constexpr int KSTEPS = K / 32;
    __shared__ _Float16 WT[F * K];   // max 64*256*2 = 32 KB (<< 320 KB WGP LDS)

    const int tid = threadIdx.x;
    // Cooperative load + transpose + f32->f16 convert of W into LDS.
    for (int idx = tid; idx < F * K; idx += 128) {
        const int f = idx / K;            // K is a power of two -> shifts
        const int k = idx - f * K;
        WT[idx] = (_Float16)W[k * F + f];
    }
    __syncthreads();

    const int lane = tid & 31;
    const int wave = tid >> 5;
    const int row0 = blockIdx.x * 64 + wave * 16;

    // A-matrix (16x32 f16) layout: lanes 0-15 hold M=lane, K base 0;
    // lanes 16-31 hold M=lane-16, K base 8. Element j: k = kb + (j<8 ? j : j+8).
    const int arow = row0 + (lane & 15);
    int srcRow = (arow < N) ? arow : (N - 1);
    if (GATHER) srcRow = nodes[srcRow];
    const float* xrow = X + (size_t)srcRow * K;
    const int kbA = (lane < 16) ? 0 : 8;
    // B-matrix (32x16 f16) layout: lane = column, same K order as A:
    // lanes 0-15 K base 0, lanes 16-31 K base 8; element j: k = kb + (j<8 ? j : j+8).
    const int kbB = (lane < 16) ? 0 : 8;

    // ---- Load all A fragments once (reused across every n-tile) ----
    v16h afrag[KSTEPS];
    #pragma unroll
    for (int s = 0; s < KSTEPS; ++s) {
        const float* xr = xrow + s * 32 + kbA;
        float4 p0 = *(const float4*)(xr + 0);
        float4 p1 = *(const float4*)(xr + 4);
        float4 p2 = *(const float4*)(xr + 16);
        float4 p3 = *(const float4*)(xr + 20);
        v16h a;
        a[0]  = (_Float16)p0.x; a[1]  = (_Float16)p0.y;
        a[2]  = (_Float16)p0.z; a[3]  = (_Float16)p0.w;
        a[4]  = (_Float16)p1.x; a[5]  = (_Float16)p1.y;
        a[6]  = (_Float16)p1.z; a[7]  = (_Float16)p1.w;
        a[8]  = (_Float16)p2.x; a[9]  = (_Float16)p2.y;
        a[10] = (_Float16)p2.z; a[11] = (_Float16)p2.w;
        a[12] = (_Float16)p3.x; a[13] = (_Float16)p3.y;
        a[14] = (_Float16)p3.z; a[15] = (_Float16)p3.w;
        afrag[s] = a;
    }

    // ---- Per-lane dinv for the 8 output rows (invariant across n-tiles) ----
    const int rbase = row0 + ((lane >> 4) << 3);
    float dv[8];
    #pragma unroll
    for (int r = 0; r < 8; ++r) {
        const int row = rbase + r;
        dv[r] = (row < N) ? dinv[row] : 0.0f;
    }
    const bool fullTile = (row0 + 16 <= N);   // wave-uniform predicate

    for (int n0 = 0; n0 < F; n0 += 16) {
        const int col = n0 + (lane & 15);

        // ---- Issue ALL B-fragment LDS loads up front (pipelined ds loads) ----
        v16h bfrag[KSTEPS];
        #pragma unroll
        for (int s = 0; s < KSTEPS; ++s) {
            const _Float16* wp = &WT[col * K + s * 32 + kbB];
            v8h blo = *(const v8h*)(wp);
            v8h bhi = *(const v8h*)(wp + 16);
            v16h b;
            #pragma unroll
            for (int j = 0; j < 8; ++j) { b[j] = blo[j]; b[j + 8] = bhi[j]; }
            bfrag[s] = b;
        }

        // ---- WMMA chain over K ----
        v8f c = {};
        #pragma unroll
        for (int s = 0; s < KSTEPS; ++s) {
            c = __builtin_amdgcn_wmma_f32_16x16x32_f16(
                    false, afrag[s], false, bfrag[s], (short)0, c, false, false);
        }

        // C/D layout: VGPR r, lanes 0-15 -> M=r, lanes 16-31 -> M=r+8; N=lane&15.
        if (fullTile) {
            #pragma unroll
            for (int r = 0; r < 8; ++r) {
                const int row = rbase + r;
                const float gv = c[r] * dv[r];         // fold source-side norm
                G[(size_t)row * F + col]   = gv;
                ACC[(size_t)row * F + col] = gv;       // self-loop init
            }
        } else {
            #pragma unroll
            for (int r = 0; r < 8; ++r) {
                const int row = rbase + r;
                if (row < N) {
                    const float gv = c[r] * dv[r];
                    G[(size_t)row * F + col]   = gv;
                    ACC[(size_t)row * F + col] = gv;
                }
            }
        }
    }
}

// ---------------------------------------------------------------------------
// Edge scatter: ACC[dst] += G[src], one thread per (edge, feature).
// L2-resident (working set << 192 MB); hardware f32 atomics.
// ---------------------------------------------------------------------------
template<int LOGF>
__global__ void gcn_scatter_kernel(const int* __restrict__ src, const int* __restrict__ dst,
                                   const float* __restrict__ G, float* __restrict__ ACC, int E)
{
    const long long total = (long long)E << LOGF;
    const long long tid = (long long)blockIdx.x * blockDim.x + threadIdx.x;
    if (tid >= total) return;
    const int e = (int)(tid >> LOGF);
    const int f = (int)tid & ((1 << LOGF) - 1);
    const float v = G[((size_t)src[e] << LOGF) + f];
    unsafeAtomicAdd(&ACC[((size_t)dst[e] << LOGF) + f], v);
}

// ---------------------------------------------------------------------------
// Finalize: out = relu(ACC * dinv[v] + b)
// ---------------------------------------------------------------------------
template<int LOGF>
__global__ void gcn_finalize_kernel(const float* __restrict__ ACC, const float* __restrict__ dinv,
                                    const float* __restrict__ bias, float* __restrict__ out, int N)
{
    const long long total = (long long)N << LOGF;
    const long long tid = (long long)blockIdx.x * blockDim.x + threadIdx.x;
    if (tid >= total) return;
    const int v = (int)(tid >> LOGF);
    const int f = (int)tid & ((1 << LOGF) - 1);
    float val = ACC[tid] * dinv[v] + bias[f];
    out[tid] = val > 0.0f ? val : 0.0f;
}

// ---------------------------------------------------------------------------
// Launch
// ---------------------------------------------------------------------------
extern "C" void kernel_launch(void* const* d_in, const int* in_sizes, int n_in,
                              void* d_out, int out_size, void* d_ws, size_t ws_size,
                              hipStream_t stream)
{
    const int*   nodes = (const int*)d_in[0];
    const int*   edges = (const int*)d_in[1];
    const float* emb   = (const float*)d_in[2];
    const float* W1 = (const float*)d_in[3];  const float* b1 = (const float*)d_in[4];
    const float* W2 = (const float*)d_in[5];  const float* b2 = (const float*)d_in[6];
    const float* W3 = (const float*)d_in[7];  const float* b3 = (const float*)d_in[8];

    const int N = in_sizes[0];
    const int E = in_sizes[1] / 2;
    const int* src = edges;
    const int* dst = edges + E;

    // Workspace layout (floats): dinv | G (N*64) | ACC (N*64) | X (N*64)
    const size_t Npad = ((size_t)N + 63) & ~(size_t)63;
    float* dinv = (float*)d_ws;
    float* G    = dinv + Npad;
    float* ACC  = G    + Npad * 64;
    float* X    = ACC  + Npad * 64;

    const int T = 256;
    // --- normalization ---
    deg_init_kernel<<<(N + T - 1) / T, T, 0, stream>>>(dinv, N);
    deg_accum_kernel<<<(E + T - 1) / T, T, 0, stream>>>(dst, dinv, E);
    dinv_kernel<<<(N + T - 1) / T, T, 0, stream>>>(dinv, N);

    const int gemmBlocks = (N + 63) / 64;

    // --- layer 1: 256 -> 64 (embedding gather folded into A load) ---
    gcn_gemm_kernel<256, 64, true><<<gemmBlocks, 128, 0, stream>>>(emb, W1, nodes, dinv, G, ACC, N);
    gcn_scatter_kernel<6><<<(int)(((long long)E * 64 + T - 1) / T), T, 0, stream>>>(src, dst, G, ACC, E);
    gcn_finalize_kernel<6><<<(int)(((long long)N * 64 + T - 1) / T), T, 0, stream>>>(ACC, dinv, b1, X, N);

    // --- layer 2: 64 -> 32 ---
    gcn_gemm_kernel<64, 32, false><<<gemmBlocks, 128, 0, stream>>>(X, W2, nullptr, dinv, G, ACC, N);
    gcn_scatter_kernel<5><<<(int)(((long long)E * 32 + T - 1) / T), T, 0, stream>>>(src, dst, G, ACC, E);
    gcn_finalize_kernel<5><<<(int)(((long long)N * 32 + T - 1) / T), T, 0, stream>>>(ACC, dinv, b2, X, N);

    // --- layer 3: 32 -> 16 (finalize writes d_out) ---
    gcn_gemm_kernel<32, 16, false><<<gemmBlocks, 128, 0, stream>>>(X, W3, nullptr, dinv, G, ACC, N);
    gcn_scatter_kernel<4><<<(int)(((long long)E * 16 + T - 1) / T), T, 0, stream>>>(src, dst, G, ACC, E);
    gcn_finalize_kernel<4><<<(int)(((long long)N * 16 + T - 1) / T), T, 0, stream>>>(ACC, dinv, b3, (float*)d_out, N);
}
